// CrossModalAttentionLayerWithPartialTraining_63376537420501
// MI455X (gfx1250) — compile-verified, matrix operand order checked by reading, not run on hardware
//
#include <hip/hip_runtime.h>
#include <math.h>

#define Dm   256
#define Hh   8
#define HDh  32
#define Nn   256
#define Pp   1024
#define Qq   64
#define EPSf 1e-5f

typedef __attribute__((ext_vector_type(16))) __bf16 v16bf;
typedef __attribute__((ext_vector_type(8)))  float  v8f;

// Load one 16x32 bf16 WMMA operand fragment (A-layout / B-as-W-rows layout).
// Per ISA: lane (hi,lr) needs K = [8*hi, 8*hi+8) and [16+8*hi, 16+8*hi+8),
// i.e. two contiguous 8-float chunks -> four b128 loads, then RNE cvt to bf16.
__device__ __forceinline__ v16bf load_frag(const float* __restrict__ row, int kt, int hi)
{
    const float4* p0 = (const float4*)(row + kt + 8 * hi);
    const float4* p1 = (const float4*)(row + kt + 16 + 8 * hi);
    float4 c0 = p0[0], c1 = p0[1], c2 = p1[0], c3 = p1[1];
    v16bf f;
    f[0]  = (__bf16)c0.x; f[1]  = (__bf16)c0.y; f[2]  = (__bf16)c0.z; f[3]  = (__bf16)c0.w;
    f[4]  = (__bf16)c1.x; f[5]  = (__bf16)c1.y; f[6]  = (__bf16)c1.z; f[7]  = (__bf16)c1.w;
    f[8]  = (__bf16)c2.x; f[9]  = (__bf16)c2.y; f[10] = (__bf16)c2.z; f[11] = (__bf16)c2.w;
    f[12] = (__bf16)c3.x; f[13] = (__bf16)c3.y; f[14] = (__bf16)c3.z; f[15] = (__bf16)c3.w;
    return f;
}

// ---------------------------------------------------------------------------
// Stage 1: fused per-expert single-query cross attention (memory-bound: streams
// the 268MB of expert weights exactly once; patches stay L2-resident).
// block = expert n, 256 threads (8 waves).
// ---------------------------------------------------------------------------
__global__ __launch_bounds__(256)
void expert_attn_kernel(const float* __restrict__ patches,   // [P, D]
                        const float* __restrict__ tokens,    // [N, D]
                        const float* __restrict__ Wqkv,      // [N, 3D, D]
                        const float* __restrict__ bqkv,      // [N, 3D]
                        const float* __restrict__ Wo,        // [N, D, D]
                        const float* __restrict__ bo,        // [N, D]
                        const float* __restrict__ g,
                        const float* __restrict__ bb,
                        float* __restrict__ AI,              // [N, D] (post-LN)
                        float* __restrict__ att_out)         // [N, P]
{
    const int n   = blockIdx.x;
    const int tid = threadIdx.x;
    const float* We = Wqkv + (size_t)n * (3 * Dm) * Dm;
    const float* be = bqkv + (size_t)n * (3 * Dm);

    __shared__ __align__(16) float t_s[Dm];
    __shared__ __align__(16) float q_s[Dm];
    __shared__ __align__(16) float r_s[Hh][Dm];
    __shared__ __align__(16) float s_s[Hh][Pp];
    __shared__ __align__(16) float m_s[Hh][Dm];
    __shared__ __align__(16) float c_s[Dm];
    __shared__ float red[16];

    t_s[tid] = tokens[(size_t)n * Dm + tid];
    __syncthreads();

    // q[e] = Wq_n[e,:] . t + bq[e]        (row-per-thread streaming)
    {
        const float4* wr = (const float4*)(We + (size_t)tid * Dm);
        const float4* tv = (const float4*)t_s;
        float acc = 0.f;
        #pragma unroll 4
        for (int i = 0; i < Dm / 4; ++i) {
            float4 w = wr[i]; float4 x = tv[i];
            acc += w.x * x.x + w.y * x.y + w.z * x.z + w.w * x.w;
        }
        q_s[tid] = acc + be[tid];
    }
    __syncthreads();

    // r[h][d] = sum_j q[h*32+j] * Wk[h*32+j][d]   (coalesced across d=tid)
    #pragma unroll
    for (int h = 0; h < Hh; ++h) {
        float acc = 0.f;
        for (int j = 0; j < HDh; ++j)
            acc += q_s[h * HDh + j] * We[(size_t)(Dm + h * HDh + j) * Dm + tid];
        r_s[h][tid] = acc;
    }
    __syncthreads();

    // scores s[h][p] = (patch_p . r_h) / sqrt(HD)
    const float scale = 0.17677669529663687f; // 1/sqrt(32)
    for (int p = tid; p < Pp; p += 256) {
        const float4* pr = (const float4*)(patches + (size_t)p * Dm);
        float acc[Hh] = {};
        for (int i = 0; i < Dm / 4; ++i) {
            float4 x = pr[i];
            #pragma unroll
            for (int h = 0; h < Hh; ++h) {
                float4 rv = ((const float4*)r_s[h])[i];
                acc[h] += x.x * rv.x + x.y * rv.y + x.z * rv.z + x.w * rv.w;
            }
        }
        #pragma unroll
        for (int h = 0; h < Hh; ++h) s_s[h][p] = acc[h] * scale;
    }
    __syncthreads();

    // softmax over P, one wave per head
    {
        int h = tid >> 5, lane = tid & 31;
        float mx = -1e30f;
        for (int p = lane; p < Pp; p += 32) mx = fmaxf(mx, s_s[h][p]);
        for (int off = 16; off; off >>= 1) mx = fmaxf(mx, __shfl_xor(mx, off, 32));
        float sum = 0.f;
        for (int p = lane; p < Pp; p += 32) {
            float e = __expf(s_s[h][p] - mx);
            s_s[h][p] = e; sum += e;
        }
        for (int off = 16; off; off >>= 1) sum += __shfl_xor(sum, off, 32);
        float inv = 1.f / sum;
        for (int p = lane; p < Pp; p += 32) s_s[h][p] *= inv;
    }
    __syncthreads();

    // head-averaged attention weights -> output
    for (int p = tid; p < Pp; p += 256) {
        float a = 0.f;
        #pragma unroll
        for (int h = 0; h < Hh; ++h) a += s_s[h][p];
        att_out[(size_t)n * Pp + p] = a * (1.0f / Hh);
    }

    // m[h][d] = sum_p att[h][p] * patch[p][d]   (coalesced across d=tid)
    {
        float acc[Hh] = {};
        for (int p = 0; p < Pp; ++p) {
            float x = patches[(size_t)p * Dm + tid];
            #pragma unroll
            for (int h = 0; h < Hh; ++h) acc[h] += s_s[h][p] * x;
        }
        #pragma unroll
        for (int h = 0; h < Hh; ++h) m_s[h][tid] = acc[h];
    }
    __syncthreads();

    // ctx[e] = Wv_n[e,:] . m[h(e)] + bv[e]
    {
        int h = tid >> 5;
        const float4* wr = (const float4*)(We + (size_t)(2 * Dm + tid) * Dm);
        const float4* mv = (const float4*)m_s[h];
        float acc = 0.f;
        for (int i = 0; i < Dm / 4; ++i) {
            float4 w = wr[i]; float4 x = mv[i];
            acc += w.x * x.x + w.y * x.y + w.z * x.z + w.w * x.w;
        }
        c_s[tid] = acc + be[2 * Dm + tid];
    }
    __syncthreads();

    // attended[e] = Wo_n[e,:] . ctx + bo[e]; residual + LayerNorm
    {
        const float4* wr = (const float4*)(Wo + (size_t)n * Dm * Dm + (size_t)tid * Dm);
        const float4* cv = (const float4*)c_s;
        float acc = 0.f;
        for (int i = 0; i < Dm / 4; ++i) {
            float4 w = wr[i]; float4 x = cv[i];
            acc += w.x * x.x + w.y * x.y + w.z * x.z + w.w * x.w;
        }
        float yv = acc + bo[(size_t)n * Dm + tid] + t_s[tid];
        float s1 = yv, s2 = yv * yv;
        for (int off = 16; off; off >>= 1) {
            s1 += __shfl_xor(s1, off, 32);
            s2 += __shfl_xor(s2, off, 32);
        }
        int wave = tid >> 5, lane = tid & 31;
        if (lane == 0) { red[wave] = s1; red[8 + wave] = s2; }
        __syncthreads();
        float t1 = 0.f, t2 = 0.f;
        #pragma unroll
        for (int w = 0; w < 8; ++w) { t1 += red[w]; t2 += red[8 + w]; }
        float mean = t1 * (1.f / Dm);
        float var  = t2 * (1.f / Dm) - mean * mean;
        AI[(size_t)n * Dm + tid] = (yv - mean) * rsqrtf(var + EPSf) * g[tid] + bb[tid];
    }
}

// ---------------------------------------------------------------------------
// WMMA GEMM: C[M,N] = A[M,K] @ W[N,K]^T + bias (+ optional ReLU).
// One wave per 32x32 C block (2x2 tiles of 16x16), bf16 A/B, f32 accumulate,
// v_wmma_f32_16x16x32_bf16. Requires M,N % 32 == 0 and K % 64 == 0.
// ---------------------------------------------------------------------------
__global__ __launch_bounds__(32)
void gemm_wmma_kernel(const float* __restrict__ A, const float* __restrict__ W,
                      const float* __restrict__ bias, float* __restrict__ C,
                      int M, int N, int K, int relu)
{
    const int bm   = blockIdx.x * 32;
    const int bn   = blockIdx.y * 32;
    const int lane = threadIdx.x & 31;
    const int hi   = lane >> 4;          // which 16-lane half
    const int lr   = lane & 15;

    const float* a0 = A + (size_t)(bm + lr) * K;
    const float* a1 = A + (size_t)(bm + 16 + lr) * K;
    const float* b0 = W + (size_t)(bn + lr) * K;
    const float* b1 = W + (size_t)(bn + 16 + lr) * K;

    v8f c00 = {}, c01 = {}, c10 = {}, c11 = {};
    for (int kt = 0; kt < K; kt += 64) {
        #pragma unroll
        for (int u = 0; u < 2; ++u) {
            const int k = kt + 32 * u;
            v16bf fa0 = load_frag(a0, k, hi);
            v16bf fa1 = load_frag(a1, k, hi);
            v16bf fb0 = load_frag(b0, k, hi);
            v16bf fb1 = load_frag(b1, k, hi);
            c00 = __builtin_amdgcn_wmma_f32_16x16x32_bf16(false, fa0, false, fb0, (short)0, c00, false, false);
            c01 = __builtin_amdgcn_wmma_f32_16x16x32_bf16(false, fa0, false, fb1, (short)0, c01, false, false);
            c10 = __builtin_amdgcn_wmma_f32_16x16x32_bf16(false, fa1, false, fb0, (short)0, c10, false, false);
            c11 = __builtin_amdgcn_wmma_f32_16x16x32_bf16(false, fa1, false, fb1, (short)0, c11, false, false);
        }
    }

    // C/D layout: N = lane&15 (+16 per n-tile), M = v + 8*half (+16 per m-tile)
    const int n0 = bn + lr;
    const int n1 = bn + 16 + lr;
    const float bv0 = bias ? bias[n0] : 0.f;
    const float bv1 = bias ? bias[n1] : 0.f;
    #pragma unroll
    for (int v = 0; v < 8; ++v) {
        int m0 = bm + v + 8 * hi;
        int m1 = m0 + 16;
        float o00 = c00[v] + bv0, o01 = c01[v] + bv1;
        float o10 = c10[v] + bv0, o11 = c11[v] + bv1;
        if (relu) {
            o00 = fmaxf(o00, 0.f); o01 = fmaxf(o01, 0.f);
            o10 = fmaxf(o10, 0.f); o11 = fmaxf(o11, 0.f);
        }
        C[(size_t)m0 * N + n0] = o00;
        C[(size_t)m0 * N + n1] = o01;
        C[(size_t)m1 * N + n0] = o10;
        C[(size_t)m1 * N + n1] = o11;
    }
}

// ---------------------------------------------------------------------------
// Multi-head attention core (Sq == 256, one row per thread, one head per
// block). Online softmax; per-head K/V staged in dynamic LDS.
// ---------------------------------------------------------------------------
__global__ __launch_bounds__(256)
void attn_kernel(const float* __restrict__ q, const float* __restrict__ k,
                 const float* __restrict__ v, float* __restrict__ ctx,
                 int Sk, float scale)
{
    extern __shared__ float smem[];
    float* kh = smem;
    float* vh = smem + (size_t)Sk * HDh;
    const int h = blockIdx.x, tid = threadIdx.x;

    for (int idx = tid; idx < Sk * HDh; idx += 256) {
        int row = idx >> 5, col = idx & 31;
        kh[idx] = k[(size_t)row * Dm + h * HDh + col];
        vh[idx] = v[(size_t)row * Dm + h * HDh + col];
    }
    __syncthreads();

    float qr[HDh];
    #pragma unroll
    for (int j = 0; j < HDh; ++j) qr[j] = q[(size_t)tid * Dm + h * HDh + j];

    float mx = -1e30f, sum = 0.f;
    float acc[HDh] = {};
    for (int kk = 0; kk < Sk; ++kk) {
        float s = 0.f;
        #pragma unroll
        for (int j = 0; j < HDh; ++j) s += qr[j] * kh[kk * HDh + j];
        s *= scale;
        float nm = fmaxf(mx, s);
        float corr = __expf(mx - nm);
        float w = __expf(s - nm);
        sum = sum * corr + w;
        #pragma unroll
        for (int j = 0; j < HDh; ++j) acc[j] = acc[j] * corr + w * vh[kk * HDh + j];
        mx = nm;
    }
    float inv = 1.f / sum;
    #pragma unroll
    for (int j = 0; j < HDh; ++j) ctx[(size_t)tid * Dm + h * HDh + j] = acc[j] * inv;
}

// ---------------------------------------------------------------------------
// out = LayerNorm(x + res) * g + b, one 256-wide row per block.
// ---------------------------------------------------------------------------
__global__ __launch_bounds__(256)
void lnres_kernel(const float* __restrict__ x, const float* __restrict__ res,
                  const float* __restrict__ g, const float* __restrict__ b,
                  float* __restrict__ out)
{
    const int row = blockIdx.x, tid = threadIdx.x;
    __shared__ float red[16];
    float yv = x[(size_t)row * Dm + tid] + res[(size_t)row * Dm + tid];
    float s1 = yv, s2 = yv * yv;
    for (int off = 16; off; off >>= 1) {
        s1 += __shfl_xor(s1, off, 32);
        s2 += __shfl_xor(s2, off, 32);
    }
    int wave = tid >> 5, lane = tid & 31;
    if (lane == 0) { red[wave] = s1; red[8 + wave] = s2; }
    __syncthreads();
    float t1 = 0.f, t2 = 0.f;
    #pragma unroll
    for (int w = 0; w < 8; ++w) { t1 += red[w]; t2 += red[8 + w]; }
    float mean = t1 * (1.f / Dm);
    float var  = t2 * (1.f / Dm) - mean * mean;
    out[(size_t)row * Dm + tid] = (yv - mean) * rsqrtf(var + EPSf) * g[tid] + b[tid];
}

// ---------------------------------------------------------------------------
extern "C" void kernel_launch(void* const* d_in, const int* in_sizes, int n_in,
                              void* d_out, int out_size, void* d_ws, size_t ws_size,
                              hipStream_t stream)
{
    const float* patches = (const float*)d_in[0];
    const float* tokens  = (const float*)d_in[1];
    const float* QE      = (const float*)d_in[2];
    const float* Wqkv_e  = (const float*)d_in[3];
    const float* bqkv_e  = (const float*)d_in[4];
    const float* Wo_e    = (const float*)d_in[5];
    const float* bo_e    = (const float*)d_in[6];
    const float* Wqkv_q  = (const float*)d_in[7];
    const float* bqkv_q  = (const float*)d_in[8];
    const float* Wo_q    = (const float*)d_in[9];
    const float* bo_q    = (const float*)d_in[10];
    const float* Wqkv_s  = (const float*)d_in[11];
    const float* bqkv_s  = (const float*)d_in[12];
    const float* Wo_s    = (const float*)d_in[13];
    const float* bo_s    = (const float*)d_in[14];
    const float* W1      = (const float*)d_in[15];
    const float* b1      = (const float*)d_in[16];
    const float* W2      = (const float*)d_in[17];
    const float* b2      = (const float*)d_in[18];
    const float* ln1_g   = (const float*)d_in[19];
    const float* ln1_b   = (const float*)d_in[20];
    const float* ln2_g   = (const float*)d_in[21];
    const float* ln2_b   = (const float*)d_in[22];
    const float* ln3_g   = (const float*)d_in[23];
    const float* ln3_b   = (const float*)d_in[24];

    float* out_feat = (float*)d_out;                 // [N, D]
    float* out_att  = out_feat + Nn * Dm;            // [N, P]

    float* ws = (float*)d_ws;
    float* AI = ws;                 // [256,256]
    float* qb = AI + 65536;         // [256,256]
    float* kb = qb + 65536;         // [<=256,256]
    float* vb = kb + 65536;
    float* cb = vb + 65536;
    float* pj = cb + 65536;
    float* AQ = pj + 65536;
    float* h1 = AQ + 65536;         // [256,1024]
    float* ER = h1 + 262144;        // [256,256]

    const float scale = 0.17677669529663687f;  // 1/sqrt(HD)

    // Stage 1: per-expert cross attention (HBM-bound weight streaming)
    expert_attn_kernel<<<Nn, 256, 0, stream>>>(patches, tokens, Wqkv_e, bqkv_e,
                                               Wo_e, bo_e, ln1_g, ln1_b, AI, out_att);

    // Stage 2: query cross attention  (q from AI, k/v from query_embeddings)
    gemm_wmma_kernel<<<dim3(8, 8), 32, 0, stream>>>(AI, Wqkv_q,           bqkv_q,       qb, 256, 256, 256, 0);
    gemm_wmma_kernel<<<dim3(2, 8), 32, 0, stream>>>(QE, Wqkv_q + 256*256, bqkv_q + 256, kb,  64, 256, 256, 0);
    gemm_wmma_kernel<<<dim3(2, 8), 32, 0, stream>>>(QE, Wqkv_q + 512*256, bqkv_q + 512, vb,  64, 256, 256, 0);
    attn_kernel<<<Hh, 256, (size_t)Qq * HDh * 4 * 2, stream>>>(qb, kb, vb, cb, Qq, scale);
    gemm_wmma_kernel<<<dim3(8, 8), 32, 0, stream>>>(cb, Wo_q, bo_q, pj, 256, 256, 256, 0);
    lnres_kernel<<<Nn, 256, 0, stream>>>(pj, AI, ln2_g, ln2_b, AQ);

    // Stage 3: FFN (no residual)
    gemm_wmma_kernel<<<dim3(8, 32), 32, 0, stream>>>(AQ, W1, b1, h1, 256, 1024, 256, 1);
    gemm_wmma_kernel<<<dim3(8, 8), 32, 0, stream>>>(h1, W2, b2, ER, 256, 256, 1024, 0);

    // Stage 4: expert self attention + residual + LN
    gemm_wmma_kernel<<<dim3(8, 8), 32, 0, stream>>>(ER, Wqkv_s,           bqkv_s,       qb, 256, 256, 256, 0);
    gemm_wmma_kernel<<<dim3(8, 8), 32, 0, stream>>>(ER, Wqkv_s + 256*256, bqkv_s + 256, kb, 256, 256, 256, 0);
    gemm_wmma_kernel<<<dim3(8, 8), 32, 0, stream>>>(ER, Wqkv_s + 512*256, bqkv_s + 512, vb, 256, 256, 256, 0);
    attn_kernel<<<Hh, 256, (size_t)Nn * HDh * 4 * 2, stream>>>(qb, kb, vb, cb, Nn, scale);
    gemm_wmma_kernel<<<dim3(8, 8), 32, 0, stream>>>(cb, Wo_s, bo_s, pj, 256, 256, 256, 0);
    lnres_kernel<<<Nn, 256, 0, stream>>>(pj, ER, ln3_g, ln3_b, out_feat);
}
